// Model_8117488189561
// MI455X (gfx1250) — matrix operand order repaired and checked by reading, not compiled, not run on hardware
//
#include <hip/hip_runtime.h>
#include <hip/hip_bf16.h>
#include <math.h>

// ---------------------------------------------------------------------------
// Autoformer forward for MI455X (gfx1250, wave32, WMMA).
// GEMMs + autocorrelation Gram-diagonal reduction run on v_wmma_f32_16x16x32_bf16.
// LDS tiles are laid out so every WMMA fragment is fetched with ds_load_b128.
// ---------------------------------------------------------------------------

typedef __attribute__((ext_vector_type(16))) __bf16 v16bf;
typedef __attribute__((ext_vector_type(8)))  __bf16 v8bf;
typedef __attribute__((ext_vector_type(4)))  __bf16 v4bf;
typedef __attribute__((ext_vector_type(8)))  float  v8f;
typedef __attribute__((ext_vector_type(4)))  float  v4f;

#define BB    32
#define LSEQ  3072
#define DM    128
#define DFF   512
#define CIN   8
#define COUT  8
#define NMARK 4
#define PRED  1536
#define TOPK  8

// Assemble a 16x32 A-fragment for lane (m, kh) from row-major LDS tile sA[64][32].
__device__ __forceinline__ v16bf load_a_frag(const __bf16* sA, int row, int kh)
{
    const v8bf lo = *(const v8bf*)(sA + row * 32 + kh * 8);
    const v8bf hi = *(const v8bf*)(sA + row * 32 + 16 + kh * 8);
    return __builtin_shufflevector(lo, hi, 0, 1, 2, 3, 4, 5, 6, 7,
                                           8, 9, 10, 11, 12, 13, 14, 15);
}

// Load a 32x16 B-fragment for lane (n, kh) from K-transposed LDS tile sBT[64][32].
__device__ __forceinline__ v16bf load_b_frag(const __bf16* sBT, int nrow, int kh)
{
    const v8bf lo = *(const v8bf*)(sBT + nrow * 32 + kh * 8);
    const v8bf hi = *(const v8bf*)(sBT + nrow * 32 + 16 + kh * 8);
    return __builtin_shufflevector(lo, hi, 0, 1, 2, 3, 4, 5, 6, 7,
                                           8, 9, 10, 11, 12, 13, 14, 15);
}

// ---------------------------------------------------------------------------
// Generic WMMA GEMM: C[M,N] = act( A[M,K] @ B[K,N] + bias ) + R
//   block = 256 threads (8 waves), 64x64 tile, K staged 32 wide through LDS.
// ---------------------------------------------------------------------------
__global__ void __launch_bounds__(256)
gemm_wmma_kernel(const float* __restrict__ A, const float* __restrict__ Bw,
                 const float* __restrict__ bias, const float* __restrict__ R,
                 float* __restrict__ C, int M, int N, int K, int gelu_flag)
{
    __shared__ __attribute__((aligned(32))) __bf16 sA[64 * 32];   // [m][k]
    __shared__ __attribute__((aligned(32))) __bf16 sBT[64 * 32];  // [n][k]

    const int row0 = blockIdx.x * 64;
    const int col0 = blockIdx.y * 64;
    const int tid  = threadIdx.x;
    const int wave = tid >> 5;
    const int lane = tid & 31;
    const int tr   = wave >> 1;          // row tile 0..3
    const int tcp  = (wave & 1) * 2;     // col tile pair base: 0 or 2

    v8f acc0 = {};
    v8f acc1 = {};

    const int m  = lane & 15;
    const int kh = lane >> 4;

    for (int k0 = 0; k0 < K; k0 += 32) {
        // stage A 64x32 (fp32 -> bf16), vectorized: float4 in, 8B out
        for (int i = tid; i < 64 * 32 / 4; i += 256) {
            int e = i * 4;
            int r = e >> 5, c = e & 31;
            v4f f = *(const v4f*)&A[(size_t)(row0 + r) * K + (k0 + c)];
            v4bf h;
            h[0] = (__bf16)f[0]; h[1] = (__bf16)f[1];
            h[2] = (__bf16)f[2]; h[3] = (__bf16)f[3];
            *(v4bf*)&sA[r * 32 + c] = h;
        }
        // stage B 32x64 transposed into sBT[n][k]; vectorize the LDS store along K
        for (int i = tid; i < 32 * 64 / 4; i += 256) {
            int e = i * 4;
            int r4 = e & 31, c = e >> 5;          // r4: k chunk, c: column
            v4bf h;
#pragma unroll
            for (int t = 0; t < 4; ++t)
                h[t] = (__bf16)Bw[(size_t)(k0 + r4 + t) * N + (col0 + c)];
            *(v4bf*)&sBT[c * 32 + r4] = h;
        }
        if (k0 + 32 < K)  // gfx1250 global_prefetch_b8 hint for next A chunk
            __builtin_prefetch(&A[(size_t)(row0 + (tid >> 2)) * K + k0 + 32], 0, 1);
        __syncthreads();

        const v16bf af = load_a_frag(sA, tr * 16 + m, kh);
        {
            const v16bf b0 = load_b_frag(sBT, tcp * 16 + m, kh);
            acc0 = __builtin_amdgcn_wmma_f32_16x16x32_bf16(false, af, false, b0,
                                                           (short)0, acc0, false, false);
            const v16bf b1 = load_b_frag(sBT, (tcp + 1) * 16 + m, kh);
            acc1 = __builtin_amdgcn_wmma_f32_16x16x32_bf16(false, af, false, b1,
                                                           (short)0, acc1, false, false);
        }
        __syncthreads();
    }

    // epilogue: bias -> gelu -> +R
    const int nn = lane & 15;
    const int mb = kh * 8;
#pragma unroll
    for (int t = 0; t < 2; ++t) {
        const v8f acc = t ? acc1 : acc0;
        const int gn  = col0 + (tcp + t) * 16 + nn;
#pragma unroll
        for (int r = 0; r < 8; ++r) {
            const int gm = row0 + tr * 16 + mb + r;
            float val = acc[r];
            if (bias) val += bias[gn];
            if (gelu_flag) val = 0.5f * val * (1.0f + erff(val * 0.70710678118654752f));
            if (R) val += R[(size_t)gm * N + gn];
            C[(size_t)gm * N + gn] = val;
        }
    }
}

// ---------------------------------------------------------------------------
// Autocorrelation via WMMA: corr[b,tau] += (1/128) * sum over diagonal tau of
// Gram matrix  Mgram[i,j] = <Q[b,i,:], K[b,j,:]>  (tau = (i-j) mod L).
// Each block: one 64x64 Gram tile, diagonal-binned in LDS, atomics to global.
// ---------------------------------------------------------------------------
__global__ void __launch_bounds__(256)
corr_diag_wmma_kernel(const float* __restrict__ Q, const float* __restrict__ Kt,
                      float* __restrict__ corr)
{
    __shared__ __attribute__((aligned(32))) __bf16 sA[64 * 32];   // Q tile [i][k]
    __shared__ __attribute__((aligned(32))) __bf16 sBT[64 * 32];  // K tile [j][k]
    __shared__ float sdiag[127];

    const int i0 = blockIdx.x * 64;
    const int j0 = blockIdx.y * 64;
    const int b  = blockIdx.z;
    const int tid  = threadIdx.x;
    const int wave = tid >> 5;
    const int lane = tid & 31;
    const int tr   = wave >> 1;
    const int tcp  = (wave & 1) * 2;
    const int m  = lane & 15;
    const int kh = lane >> 4;

    const float* Qb = Q  + (size_t)b * LSEQ * DM;
    const float* Kb = Kt + (size_t)b * LSEQ * DM;

    if (tid < 127) sdiag[tid] = 0.0f;

    v8f acc0 = {};
    v8f acc1 = {};

    for (int k0 = 0; k0 < DM; k0 += 32) {
        for (int i = tid; i < 64 * 32 / 4; i += 256) {
            int e = i * 4;
            int r = e >> 5, c = e & 31;
            v4f f = *(const v4f*)&Qb[(size_t)(i0 + r) * DM + (k0 + c)];
            v4bf h;
            h[0] = (__bf16)f[0]; h[1] = (__bf16)f[1];
            h[2] = (__bf16)f[2]; h[3] = (__bf16)f[3];
            *(v4bf*)&sA[r * 32 + c] = h;
        }
        // K tile is consumed as B^T, and global K rows are already [j][k]:
        // coalesced float4 read, contiguous v4bf store.
        for (int i = tid; i < 64 * 32 / 4; i += 256) {
            int e = i * 4;
            int j = e >> 5, kk = e & 31;
            v4f f = *(const v4f*)&Kb[(size_t)(j0 + j) * DM + (k0 + kk)];
            v4bf h;
            h[0] = (__bf16)f[0]; h[1] = (__bf16)f[1];
            h[2] = (__bf16)f[2]; h[3] = (__bf16)f[3];
            *(v4bf*)&sBT[j * 32 + kk] = h;
        }
        __syncthreads();

        const v16bf af = load_a_frag(sA, tr * 16 + m, kh);
        {
            const v16bf b0 = load_b_frag(sBT, tcp * 16 + m, kh);
            acc0 = __builtin_amdgcn_wmma_f32_16x16x32_bf16(false, af, false, b0,
                                                           (short)0, acc0, false, false);
            const v16bf b1 = load_b_frag(sBT, (tcp + 1) * 16 + m, kh);
            acc1 = __builtin_amdgcn_wmma_f32_16x16x32_bf16(false, af, false, b1,
                                                           (short)0, acc1, false, false);
        }
        __syncthreads();
    }

    // fold 64x64 tile into its 127 diagonals
    const int nn = lane & 15;
    const int mb = kh * 8;
#pragma unroll
    for (int t = 0; t < 2; ++t) {
        const v8f acc = t ? acc1 : acc0;
        const int dj  = (tcp + t) * 16 + nn;
#pragma unroll
        for (int r = 0; r < 8; ++r) {
            const int di = tr * 16 + mb + r;
            atomicAdd(&sdiag[di - dj + 63], acc[r]);
        }
    }
    __syncthreads();

    if (tid < 127) {
        int tau = i0 - j0 + tid - 63;
        tau = ((tau % LSEQ) + LSEQ) % LSEQ;
        atomicAdd(&corr[(size_t)b * LSEQ + tau], sdiag[tid] * (1.0f / 128.0f));
    }
}

// ---------------------------------------------------------------------------
// Elementwise / reduction kernels
// ---------------------------------------------------------------------------
__global__ void zero_kernel(float* p, int n)
{
    int i = blockIdx.x * 256 + threadIdx.x;
    if (i < n) p[i] = 0.0f;
}

// circular conv1d(width 3, CIN->DM) embedding + mark @ w_time
__global__ void embed_kernel(const float* __restrict__ x, const float* __restrict__ mark,
                             const float* __restrict__ wconv, const float* __restrict__ wtime,
                             float* __restrict__ out)
{
    int idx = blockIdx.x * 256 + threadIdx.x;
    int total = BB * LSEQ * DM;
    if (idx >= total) return;
    int n = idx & (DM - 1);
    int bl = idx >> 7;
    int l = bl % LSEQ, b = bl / LSEQ;
    float s = 0.0f;
#pragma unroll
    for (int t = 0; t < 3; ++t) {
        int ll = l + t - 1;
        if (ll < 0) ll += LSEQ;
        if (ll >= LSEQ) ll -= LSEQ;
        const float* xr = x + ((size_t)b * LSEQ + ll) * CIN;
#pragma unroll
        for (int c = 0; c < CIN; ++c)
            s += xr[c] * wconv[(t * CIN + c) * DM + n];
    }
    const float* mr = mark + ((size_t)b * LSEQ + l) * NMARK;
#pragma unroll
    for (int q = 0; q < NMARK; ++q)
        s += mr[q] * wtime[q * DM + n];
    out[idx] = s;
}

// series_decomp: seasonal = x - movavg25(x); optional mean store/accumulate
__global__ void decomp_kernel(const float* __restrict__ x, float* __restrict__ seas,
                              float* __restrict__ meanout, int C, int accum)
{
    int idx = blockIdx.x * 256 + threadIdx.x;
    int total = BB * LSEQ * C;
    if (idx >= total) return;
    int c = idx % C;
    int l = (idx / C) % LSEQ;
    int b = idx / (C * LSEQ);
    float s = 0.0f;
#pragma unroll
    for (int j = -12; j <= 12; ++j) {
        int ll = l + j;
        ll = ll < 0 ? 0 : (ll >= LSEQ ? LSEQ - 1 : ll);
        s += x[((size_t)b * LSEQ + ll) * C + c];
    }
    float mn = s * (1.0f / 25.0f);
    seas[idx] = x[idx] - mn;
    if (meanout) {
        if (accum) meanout[idx] += mn;
        else       meanout[idx]  = mn;
    }
}

// top-8 + softmax per batch over corr[b, 0..L)
__global__ void topk_softmax_kernel(const float* __restrict__ corr,
                                    float* __restrict__ wout, int* __restrict__ dout)
{
    const int b = blockIdx.x;
    const int tid = threadIdx.x;
    __shared__ float svals[256];
    __shared__ int   sidx[256];
    __shared__ int   chosen[TOPK];
    __shared__ float chosenv[TOPK];
    const float* c = corr + (size_t)b * LSEQ;

    for (int it = 0; it < TOPK; ++it) {
        float best = -1e30f; int bi = 0;
        for (int i = tid; i < LSEQ; i += 256) {
            bool skip = false;
            for (int j = 0; j < it; ++j) if (chosen[j] == i) skip = true;
            float v = c[i];
            if (!skip && v > best) { best = v; bi = i; }
        }
        svals[tid] = best; sidx[tid] = bi;
        __syncthreads();
        for (int s = 128; s > 0; s >>= 1) {
            if (tid < s && svals[tid + s] > svals[tid]) {
                svals[tid] = svals[tid + s];
                sidx[tid]  = sidx[tid + s];
            }
            __syncthreads();
        }
        if (tid == 0) { chosen[it] = sidx[0]; chosenv[it] = svals[0]; }
        __syncthreads();
    }
    if (tid == 0) {
        float mx = chosenv[0];
        float e[TOPK], s = 0.0f;
        for (int i = 0; i < TOPK; ++i) { e[i] = expf(chosenv[i] - mx); s += e[i]; }
        float inv = 1.0f / s;
        for (int i = 0; i < TOPK; ++i) {
            wout[b * TOPK + i] = e[i] * inv;
            dout[b * TOPK + i] = chosen[i];
        }
    }
}

// out[b,l,:] = sum_i w[b,i] * V[b,(l+d[b,i]) mod L, :]
__global__ void gather_combine_kernel(const float* __restrict__ V,
                                      const float* __restrict__ w, const int* __restrict__ d,
                                      float* __restrict__ out)
{
    int idx = blockIdx.x * 256 + threadIdx.x;
    int total = BB * LSEQ * DM;
    if (idx >= total) return;
    int c = idx & (DM - 1);
    int bl = idx >> 7;
    int l = bl % LSEQ, b = bl / LSEQ;
    const float* wb = w + b * TOPK;
    const int*   db = d + b * TOPK;
    float s = 0.0f;
#pragma unroll
    for (int i = 0; i < TOPK; ++i) {
        int ll = l + db[i];
        if (ll >= LSEQ) ll -= LSEQ;
        s += wb[i] * V[((size_t)b * LSEQ + ll) * DM + c];
    }
    out[idx] = s;
}

// layernorm pass 1: per-row LN, accumulate column sums for seq-mean subtract
__global__ void ln1_kernel(const float* __restrict__ x, const float* __restrict__ g,
                           const float* __restrict__ bvec, float* __restrict__ out,
                           float* __restrict__ colsum)
{
    const int row = blockIdx.x;       // b*L + l
    const int c   = threadIdx.x;      // 0..127
    const int bb  = row / LSEQ;
    float v = x[(size_t)row * DM + c];
    __shared__ float red[DM];
    red[c] = v; __syncthreads();
    for (int s = 64; s > 0; s >>= 1) { if (c < s) red[c] += red[c + s]; __syncthreads(); }
    float mu = red[0] * (1.0f / DM);
    __syncthreads();
    float dv = v - mu;
    red[c] = dv * dv; __syncthreads();
    for (int s = 64; s > 0; s >>= 1) { if (c < s) red[c] += red[c + s]; __syncthreads(); }
    float var = red[0] * (1.0f / DM);
    float xh = dv * rsqrtf(var + 1e-5f) * g[c] + bvec[c];
    out[(size_t)row * DM + c] = xh;
    atomicAdd(&colsum[bb * DM + c], xh);
}

// layernorm pass 2: subtract per-(b,c) mean over sequence
__global__ void ln2_kernel(float* __restrict__ out, const float* __restrict__ colsum)
{
    int idx = blockIdx.x * 256 + threadIdx.x;
    int total = BB * LSEQ * DM;
    if (idx >= total) return;
    int c = idx & (DM - 1);
    int b = idx >> 7; b /= LSEQ;
    out[idx] -= colsum[b * DM + c] * (1.0f / LSEQ);
}

// circular conv (width 3, DM -> COUT) of trend accumulator, added into trend
__global__ void trend_conv_kernel(const float* __restrict__ tacc,
                                  const float* __restrict__ wtrend,
                                  float* __restrict__ trend)
{
    int idx = blockIdx.x * 256 + threadIdx.x;
    int total = BB * LSEQ * COUT;
    if (idx >= total) return;
    int o = idx & (COUT - 1);
    int bl = idx >> 3;
    int l = bl % LSEQ, b = bl / LSEQ;
    float s = 0.0f;
#pragma unroll
    for (int t = 0; t < 3; ++t) {
        int ll = l + t - 1;
        if (ll < 0) ll += LSEQ;
        if (ll >= LSEQ) ll -= LSEQ;
        const float* xr = tacc + ((size_t)b * LSEQ + ll) * DM;
        const float* wr = wtrend + (size_t)t * DM * COUT + o;
        for (int c = 0; c < DM; ++c)
            s += xr[c] * wr[(size_t)c * COUT];
    }
    trend[idx] += s;
}

// final: out = trend[:, -PRED:, :] + xln[:, -PRED:, :] @ w_proj + b_proj
__global__ void final_proj_kernel(const float* __restrict__ xln,
                                  const float* __restrict__ trend,
                                  const float* __restrict__ wproj,
                                  const float* __restrict__ bproj,
                                  float* __restrict__ out)
{
    int idx = blockIdx.x * 256 + threadIdx.x;
    int total = BB * PRED * COUT;
    if (idx >= total) return;
    int o = idx & (COUT - 1);
    int bp = idx >> 3;
    int p = bp % PRED, b = bp / PRED;
    int l = (LSEQ - PRED) + p;
    const float* xr = xln + ((size_t)b * LSEQ + l) * DM;
    float s = bproj[o];
    for (int c = 0; c < DM; ++c)
        s += xr[c] * wproj[c * COUT + o];
    out[idx] = trend[((size_t)b * LSEQ + l) * COUT + o] + s;
}

// ---------------------------------------------------------------------------
// Host orchestration
// ---------------------------------------------------------------------------
// d_in order (jax tree flatten, sorted dict keys):
// 0:x_enc 1:x_mark_enc 2:x_dec 3:x_mark_dec
// 4:b_proj 5:dec_emb.w_conv 6:dec_emb.w_time
// dec_layers[0].cross: 7:bk 8:bo 9:bq 10:bv 11:wk 12:wo 13:wq 14:wv
// dec_layers[0].self : 15:bk 16:bo 17:bq 18:bv 19:wk 20:wo 21:wq 22:wv
// 23:w1 24:w2 25:w_trend
// 26:dec_norm.b 27:dec_norm.g 28:enc_emb.w_conv 29:enc_emb.w_time
// enc_layers[0].attn: 30:bk 31:bo 32:bq 33:bv 34:wk 35:wo 36:wq 37:wv  38:w1 39:w2
// enc_layers[1].attn: 40..47                                           48:w1 49:w2
// 50:enc_norm.b 51:enc_norm.g 52:w_proj

extern "C" void kernel_launch(void* const* d_in, const int* in_sizes, int n_in,
                              void* d_out, int out_size, void* d_ws, size_t ws_size,
                              hipStream_t stream)
{
    (void)in_sizes; (void)n_in; (void)out_size; (void)ws_size;
    auto P = [&](int i) -> const float* { return (const float*)d_in[i]; };

    const int   Mrows = BB * LSEQ;            // 98304
    const size_t BIG  = (size_t)Mrows * DM;   // 12.58M f32

    float* XA   = (float*)d_ws;
    float* XB   = XA   + BIG;
    float* Qb   = XB   + BIG;
    float* Kb   = Qb   + BIG;
    float* Vb   = Kb   + BIG;
    float* Tb   = Vb   + BIG;
    float* ENC  = Tb   + BIG;
    float* TACC = ENC  + BIG;
    float* Hb   = Qb;                          // FFN hidden overlays Q..T (4*BIG = M*512)
    float* CORR = TACC + BIG;
    float* WT   = CORR + (size_t)BB * LSEQ;
    int*   DT   = (int*)(WT + BB * TOPK);
    float* COLS = (float*)(DT + BB * TOPK);
    float* TREND= COLS + BB * DM;

    const int totBLD = Mrows * DM;             // 12,582,912
    const int gBLD   = (totBLD + 255) / 256;

    auto zero = [&](float* p, int n) {
        zero_kernel<<<(n + 255) / 256, 256, 0, stream>>>(p, n);
    };
    auto gemm = [&](const float* A, const float* Bw, const float* bias, const float* R,
                    float* C, int Mi, int Ni, int Ki, int gelu) {
        dim3 grid(Mi / 64, Ni / 64);
        gemm_wmma_kernel<<<grid, 256, 0, stream>>>(A, Bw, bias, R, C, Mi, Ni, Ki, gelu);
    };
    auto decomp = [&](const float* x, float* seas, float* meanout, int C, int accum) {
        int n = BB * LSEQ * C;
        decomp_kernel<<<(n + 255) / 256, 256, 0, stream>>>(x, seas, meanout, C, accum);
    };
    auto attn = [&](float* xq, const float* xkv,
                    int iwq, int ibq, int iwk, int ibk, int iwv, int ibv,
                    int iwo, int ibo) {
        gemm(xq,  P(iwq), P(ibq), nullptr, Qb, Mrows, DM, DM, 0);
        gemm(xkv, P(iwk), P(ibk), nullptr, Kb, Mrows, DM, DM, 0);
        gemm(xkv, P(iwv), P(ibv), nullptr, Vb, Mrows, DM, DM, 0);
        zero(CORR, BB * LSEQ);
        dim3 gc(LSEQ / 64, LSEQ / 64, BB);
        corr_diag_wmma_kernel<<<gc, 256, 0, stream>>>(Qb, Kb, CORR);
        topk_softmax_kernel<<<BB, 256, 0, stream>>>(CORR, WT, DT);
        gather_combine_kernel<<<gBLD, 256, 0, stream>>>(Vb, WT, DT, Tb);
        gemm(Tb, P(iwo), P(ibo), xq /*residual*/, xq /*in place*/, Mrows, DM, DM, 0);
    };
    auto layernorm = [&](float* x, float* out, int ig, int ib) {
        zero(COLS, BB * DM);
        ln1_kernel<<<Mrows, DM, 0, stream>>>(x, P(ig), P(ib), out, COLS);
        ln2_kernel<<<gBLD, 256, 0, stream>>>(out, COLS);
    };

    // ---------------- Encoder ----------------
    embed_kernel<<<gBLD, 256, 0, stream>>>(P(0), P(1), P(28), P(29), XA);
    float* cur = XA; float* oth = XB;
    const int encW[2][10] = {
        // wq,bq, wk,bk, wv,bv, wo,bo, w1,w2
        {36, 32, 34, 30, 37, 33, 35, 31, 38, 39},
        {46, 42, 44, 40, 47, 43, 45, 41, 48, 49},
    };
    for (int ly = 0; ly < 2; ++ly) {
        const int* w = encW[ly];
        attn(cur, cur, w[0], w[1], w[2], w[3], w[4], w[5], w[6], w[7]);
        decomp(cur, oth, nullptr, DM, 0);                 // x = seasonal
        gemm(oth, P(w[8]), nullptr, nullptr, Hb, Mrows, DFF, DM, 1);   // gelu(x@w1)
        gemm(Hb,  P(w[9]), nullptr, oth, cur, Mrows, DM, DFF, 0);      // x + y
        decomp(cur, oth, nullptr, DM, 0);
        float* t = cur; cur = oth; oth = t;
    }
    layernorm(cur, ENC, /*g*/51, /*b*/50);                // enc_norm

    // ---------------- Decoder ----------------
    float* S8 = TREND + (size_t)Mrows * COUT;             // (B,L,8) seasonal of x_dec
    decomp(P(2), S8, TREND, COUT, 0);                     // trend init = moving mean
    embed_kernel<<<gBLD, 256, 0, stream>>>(S8, P(3), P(5), P(6), XA);
    cur = XA; oth = XB;

    // self-attention
    attn(cur, cur, /*wq*/21, /*bq*/17, /*wk*/19, /*bk*/15, /*wv*/22, /*bv*/18,
         /*wo*/20, /*bo*/16);
    decomp(cur, oth, TACC, DM, 0);                        // t1 (store)
    { float* t = cur; cur = oth; oth = t; }
    // cross-attention (kv = encoder output)
    attn(cur, ENC, /*wq*/13, /*bq*/9, /*wk*/11, /*bk*/7, /*wv*/14, /*bv*/10,
         /*wo*/12, /*bo*/8);
    decomp(cur, oth, TACC, DM, 1);                        // t2 (accumulate)
    { float* t = cur; cur = oth; oth = t; }
    // FFN
    gemm(cur, P(23), nullptr, nullptr, Hb, Mrows, DFF, DM, 1);
    gemm(Hb,  P(24), nullptr, cur, oth, Mrows, DM, DFF, 0);
    decomp(oth, cur, TACC, DM, 1);                        // t3 (accumulate); cur = seasonal
    // trend += circ_conv(t1+t2+t3, w_trend)
    {
        int n = BB * LSEQ * COUT;
        trend_conv_kernel<<<(n + 255) / 256, 256, 0, stream>>>(TACC, P(25), TREND);
    }
    // dec_norm -> oth
    layernorm(cur, oth, /*g*/27, /*b*/26);
    // output
    {
        int n = BB * PRED * COUT;
        final_proj_kernel<<<(n + 255) / 256, 256, 0, stream>>>(oth, TREND, P(52), P(4),
                                                               (float*)d_out);
    }
}
